// LAL_43482248904966
// MI455X (gfx1250) — matrix operand order repaired and checked by reading, not compiled, not run on hardware
//
#include <hip/hip_runtime.h>

// Problem constants (from reference)
#define B_    4
#define N_    4096
#define C_    512
#define K_    20
#define HID_  64
#define EPS_  1e-5f
#define SLOPE_ 0.2f

typedef float    v8f  __attribute__((ext_vector_type(8)));
typedef _Float16 v16h __attribute__((ext_vector_type(16)));
typedef _Float16 v8h  __attribute__((ext_vector_type(8)));

// ---------------------------------------------------------------------------
// Kernel 1: KNN. Each wave scans a 512-candidate chunk for 32 points
// (candidate loads are wave-uniform -> broadcast). Per-thread sorted top-20
// kept in registers, merged through LDS with early-exit (lists are sorted).
// ---------------------------------------------------------------------------
__global__ __launch_bounds__(256) void LAL_knn_kernel(const float* __restrict__ xyz,
                                                      int* __restrict__ idx) {
  __shared__ float sd[256 * K_];   // 20 KB
  __shared__ int   si[256 * K_];   // 20 KB
  const int t    = threadIdx.x;
  const int lp   = t & 31;          // local point 0..31
  const int sub  = t >> 5;          // candidate chunk 0..7 (one per wave)
  const int p    = blockIdx.x * 32 + lp;      // global point index (b*N+n)
  const int b    = p >> 12;                   // N = 4096
  const int bbase = b << 12;

  const float qx = xyz[(size_t)p * 3 + 0];
  const float qy = xyz[(size_t)p * 3 + 1];
  const float qz = xyz[(size_t)p * 3 + 2];
  const float qsq = qx * qx + qy * qy + qz * qz;

  float bd[K_];
  int   bi[K_];
#pragma unroll
  for (int k = 0; k < K_; ++k) { bd[k] = __builtin_inff(); bi[k] = 0; }

  const int m0 = sub * (N_ / 8);   // 512-wide chunk
  for (int mm = 0; mm < N_ / 8; ++mm) {
    const int m = m0 + mm;
    const float* mp = xyz + (size_t)(bbase + m) * 3;
    const float mx = mp[0], my = mp[1], mz = mp[2];
    const float msq = mx * mx + my * my + mz * mz;
    const float d = qsq + msq - 2.0f * (qx * mx + qy * my + qz * mz);
    if (d < bd[K_ - 1]) {
      bd[K_ - 1] = d; bi[K_ - 1] = m;
#pragma unroll
      for (int j = K_ - 1; j > 0; --j) {
        if (bd[j] < bd[j - 1]) {   // strict: equal dist keeps lower index first
          float td = bd[j]; bd[j] = bd[j - 1]; bd[j - 1] = td;
          int   ti = bi[j]; bi[j] = bi[j - 1]; bi[j - 1] = ti;
        }
      }
    }
  }

#pragma unroll
  for (int k = 0; k < K_; ++k) { sd[t * K_ + k] = bd[k]; si[t * K_ + k] = bi[k]; }
  __syncthreads();

  // wave 0 merges the 8 chunk-lists per point (chunks in ascending index order)
  if (t < 32) {
    for (int s = 1; s < 8; ++s) {
      const int src = (s << 5) | lp;
#pragma unroll
      for (int k = 0; k < K_; ++k) {
        const float d = sd[src * K_ + k];
        if (d >= bd[K_ - 1]) break;  // source list sorted ascending
        const int m = si[src * K_ + k];
        bd[K_ - 1] = d; bi[K_ - 1] = m;
#pragma unroll
        for (int j = K_ - 1; j > 0; --j) {
          if (bd[j] < bd[j - 1]) {
            float td = bd[j]; bd[j] = bd[j - 1]; bd[j - 1] = td;
            int   ti = bi[j]; bi[j] = bi[j - 1]; bi[j - 1] = ti;
          }
        }
      }
    }
#pragma unroll
    for (int k = 0; k < K_; ++k) idx[(size_t)p * K_ + k] = bi[k];
  }
}

// ---------------------------------------------------------------------------
// fp32 -> fp16 staging kernels
// ---------------------------------------------------------------------------
__global__ __launch_bounds__(256) void LAL_cvt_f16(const float* __restrict__ src,
                                                   _Float16* __restrict__ dst) {
  const int i = blockIdx.x * 256 + threadIdx.x;
  dst[i] = (_Float16)src[i];
}

// Weff (128 x 512) fp16: row d (d<64) = w1[d, 0:512]; row 64+d = w1[d, 512:1024]
__global__ __launch_bounds__(256) void LAL_cvt_w1(const float* __restrict__ w1,
                                                  _Float16* __restrict__ wh) {
  const int i  = blockIdx.x * 256 + threadIdx.x;   // 0 .. 128*512-1
  const int dd = i >> 9;
  const int c  = i & 511;
  wh[i] = (_Float16)w1[(dd & 63) * 1024 + ((dd >> 6) << 9) + c];
}

// ---------------------------------------------------------------------------
// Kernel 2: z = xh @ Weff^T  (16384 x 512) @ (512 x 128), f16 WMMA, f32 acc.
// A frag per lane: 16 contiguous K at 16*hi of each 32-block.
// B frag per lane: 16 contiguous K at 16*hi of each 32-block (row of Weff).
// ---------------------------------------------------------------------------
__global__ __launch_bounds__(256) void LAL_gemm1_kernel(const _Float16* __restrict__ xh,
                                                        const _Float16* __restrict__ wh,
                                                        float* __restrict__ z) {
  const int lane = threadIdx.x & 31;
  const int wave = threadIdx.x >> 5;      // 0..7
  const int r0   = blockIdx.x * 16;
  const int d0   = wave * 16;
  const int hi   = lane >> 4;
  const int l15  = lane & 15;

  const _Float16* __restrict__ aptr = xh + (size_t)(r0 + l15) * C_ + 16 * hi;
  const _Float16* __restrict__ bptr = wh + (size_t)(d0 + l15) * C_ + 16 * hi;

  v8f acc = {};
#pragma unroll
  for (int k = 0; k < C_; k += 32) {
    const v8h alo = *(const v8h*)(aptr + k);
    const v8h ahi = *(const v8h*)(aptr + k + 8);
    v16h a;
#pragma unroll
    for (int j = 0; j < 8; ++j) { a[j] = alo[j]; a[8 + j] = ahi[j]; }
    const v16h b = *(const v16h*)(bptr + k);
    acc = __builtin_amdgcn_wmma_f32_16x16x32_f16(false, a, false, b, (short)0, acc,
                                                 false, false);
  }
  const int crow = r0 + 8 * hi;
#pragma unroll
  for (int r = 0; r < 8; ++r)
    z[(size_t)(crow + r) * 128 + d0 + l15] = acc[r];
}

// ---------------------------------------------------------------------------
// Kernel 3: BN1 partial sums over all (b,n,k) for the 64 hidden channels.
// h[b,n,k,d] = z[gather,d] + z[row,64+d] - z[row,d]
// ---------------------------------------------------------------------------
#define S1_BLOCKS 1024
#define S1_ROWS   (B_ * N_ * K_)          // 327680
#define S1_RPB    (S1_ROWS / S1_BLOCKS)   // 320

__global__ __launch_bounds__(256) void LAL_stats1_partial(const float* __restrict__ z,
                                                          const int* __restrict__ idx,
                                                          float* __restrict__ part1) {
  __shared__ float ssum[256], ssq[256];
  const int t = threadIdx.x;
  const int d = t & 63;
  const int g = t >> 6;                    // 0..3
  const int rbase = blockIdx.x * S1_RPB;
  float s = 0.f, s2 = 0.f;
  for (int i = g; i < S1_RPB; i += 4) {
    const int r = rbase + i;
    const int p = r / K_;                  // b*N + n
    const int b = p >> 12;
    const int j = idx[r];
    const float h = z[(size_t)((b << 12) + j) * 128 + d]
                  + z[(size_t)p * 128 + 64 + d]
                  - z[(size_t)p * 128 + d];
    s += h; s2 += h * h;
  }
  ssum[t] = s; ssq[t] = s2;
  __syncthreads();
  if (g == 0) {
    s  = ssum[d] + ssum[64 + d] + ssum[128 + d] + ssum[192 + d];
    s2 = ssq[d]  + ssq[64 + d]  + ssq[128 + d]  + ssq[192 + d];
    part1[blockIdx.x * 128 + d]      = s;
    part1[blockIdx.x * 128 + 64 + d] = s2;
  }
}

__global__ __launch_bounds__(64) void LAL_stats1_final(const float* __restrict__ part1,
                                                       const float* __restrict__ g1,
                                                       const float* __restrict__ b1,
                                                       float* __restrict__ st1) {
  const int d = threadIdx.x;   // 0..63
  float s = 0.f, s2 = 0.f;
  for (int i = 0; i < S1_BLOCKS; ++i) {
    s  += part1[i * 128 + d];
    s2 += part1[i * 128 + 64 + d];
  }
  const float inv = 1.0f / (float)S1_ROWS;
  const float mu  = s * inv;
  float var = s2 * inv - mu * mu;
  var = var < 0.f ? 0.f : var;
  const float sc = g1[d] * rsqrtf(var + EPS_);
  st1[d]      = sc;
  st1[64 + d] = b1[d] - mu * sc;
}

// ---------------------------------------------------------------------------
// Kernel 4: normalize + leaky-relu + max over K -> hmax (B*N, 64)
// ---------------------------------------------------------------------------
__global__ __launch_bounds__(256) void LAL_hmax_kernel(const float* __restrict__ z,
                                                       const int* __restrict__ idx,
                                                       const float* __restrict__ st1,
                                                       float* __restrict__ hmax) {
  const int t = threadIdx.x;
  const int d = t & 63;
  const int p = blockIdx.x * 4 + (t >> 6);
  const int b = p >> 12;
  const float sc = st1[d], sh = st1[64 + d];
  const float tc = z[(size_t)p * 128 + 64 + d] - z[(size_t)p * 128 + d];
  float m = -__builtin_inff();
#pragma unroll
  for (int k = 0; k < K_; ++k) {
    const int j = idx[(size_t)p * K_ + k];
    float h = z[(size_t)((b << 12) + j) * 128 + d] + tc;
    h = fmaf(h, sc, sh);
    h = (h >= 0.f) ? h : SLOPE_ * h;
    m = fmaxf(m, h);
  }
  hmax[(size_t)p * 64 + d] = m;
}

// ---------------------------------------------------------------------------
// Kernel 5: y_pre = hmaxh @ w2h^T  (16384 x 64) @ (64 x 512), f16 WMMA.
// ---------------------------------------------------------------------------
__global__ __launch_bounds__(256) void LAL_gemm2_kernel(const _Float16* __restrict__ hmaxh,
                                                        const _Float16* __restrict__ w2h,
                                                        float* __restrict__ y) {
  const int lane = threadIdx.x & 31;
  const int wave = threadIdx.x >> 5;
  const int r0   = blockIdx.x * 16;
  const int e0   = blockIdx.y * 128 + wave * 16;
  const int hi   = lane >> 4;
  const int l15  = lane & 15;

  const _Float16* __restrict__ aptr = hmaxh + (size_t)(r0 + l15) * HID_ + 16 * hi;
  const _Float16* __restrict__ bptr = w2h   + (size_t)(e0 + l15) * HID_ + 16 * hi;

  v8f acc = {};
#pragma unroll
  for (int k = 0; k < HID_; k += 32) {
    const v8h alo = *(const v8h*)(aptr + k);
    const v8h ahi = *(const v8h*)(aptr + k + 8);
    v16h a;
#pragma unroll
    for (int j = 0; j < 8; ++j) { a[j] = alo[j]; a[8 + j] = ahi[j]; }
    const v16h b = *(const v16h*)(bptr + k);
    acc = __builtin_amdgcn_wmma_f32_16x16x32_f16(false, a, false, b, (short)0, acc,
                                                 false, false);
  }
  const int crow = r0 + 8 * hi;
#pragma unroll
  for (int r = 0; r < 8; ++r)
    y[(size_t)(crow + r) * C_ + e0 + l15] = acc[r];
}

// ---------------------------------------------------------------------------
// Kernel 6/7: BN2 stats (deterministic two-stage reduction over 16384 rows)
// ---------------------------------------------------------------------------
__global__ __launch_bounds__(256) void LAL_stats2_partial(const float* __restrict__ y,
                                                          float* __restrict__ part2) {
  const int t  = threadIdx.x;
  const int r0 = blockIdx.x * 32;
  float s0 = 0.f, q0 = 0.f, s1 = 0.f, q1 = 0.f;
  for (int i = 0; i < 32; ++i) {
    const float* row = y + (size_t)(r0 + i) * C_;
    const float a = row[t];
    const float c = row[t + 256];
    s0 += a; q0 += a * a;
    s1 += c; q1 += c * c;
  }
  float* pb = part2 + (size_t)blockIdx.x * 1024;
  pb[t]             = s0;
  pb[t + 256]       = s1;
  pb[512 + t]       = q0;
  pb[512 + t + 256] = q1;
}

__global__ __launch_bounds__(512) void LAL_stats2_final(const float* __restrict__ part2,
                                                        const float* __restrict__ g2,
                                                        const float* __restrict__ b2,
                                                        float* __restrict__ st2) {
  const int e = threadIdx.x;   // 0..511
  float s = 0.f, q = 0.f;
  for (int i = 0; i < 512; ++i) {
    s += part2[(size_t)i * 1024 + e];
    q += part2[(size_t)i * 1024 + 512 + e];
  }
  const float inv = 1.0f / (float)(B_ * N_);
  const float mu  = s * inv;
  float var = q * inv - mu * mu;
  var = var < 0.f ? 0.f : var;
  const float sc = g2[e] * rsqrtf(var + EPS_);
  st2[e]       = sc;
  st2[512 + e] = b2[e] - mu * sc;
}

// ---------------------------------------------------------------------------
// Kernel 8: in-place BN2 + leaky-relu on d_out
// ---------------------------------------------------------------------------
__global__ __launch_bounds__(256) void LAL_apply_bn2(float* __restrict__ y,
                                                     const float* __restrict__ st2) {
  const int gid = blockIdx.x * 256 + threadIdx.x;
  const int e = gid & (C_ - 1);
  float v = y[gid];
  v = fmaf(v, st2[e], st2[C_ + e]);
  y[gid] = (v >= 0.f) ? v : SLOPE_ * v;
}

// ---------------------------------------------------------------------------
// Host launcher
// ---------------------------------------------------------------------------
extern "C" void kernel_launch(void* const* d_in, const int* in_sizes, int n_in,
                              void* d_out, int out_size, void* d_ws, size_t ws_size,
                              hipStream_t stream) {
  (void)in_sizes; (void)n_in; (void)out_size; (void)ws_size;
  const float* x   = (const float*)d_in[0];
  const float* xyz = (const float*)d_in[1];
  const float* w1  = (const float*)d_in[2];
  const float* g1  = (const float*)d_in[3];
  const float* b1  = (const float*)d_in[4];
  const float* w2  = (const float*)d_in[5];
  const float* g2  = (const float*)d_in[6];
  const float* b2  = (const float*)d_in[7];
  float* y = (float*)d_out;

  char* ws = (char*)d_ws;
  // Workspace layout (~34 MB total)
  int*       idx    = (int*)      (ws + 0);          //  1,310,720 B
  float*     z      = (float*)    (ws + 1310720);    //  8,388,608 B
  float*     hmax   = (float*)    (ws + 9699328);    //  4,194,304 B
  float*     part1  = (float*)    (ws + 13893632);   //    524,288 B
  float*     st1    = (float*)    (ws + 14417920);   //        512 B
  float*     part2  = (float*)    (ws + 14418432);   //  2,097,152 B
  float*     st2    = (float*)    (ws + 16515584);   //      4,096 B
  _Float16*  xh     = (_Float16*) (ws + 16519680);   // 16,777,216 B
  _Float16*  wh1    = (_Float16*) (ws + 33296896);   //    131,072 B
  _Float16*  w2h    = (_Float16*) (ws + 33427968);   //     65,536 B
  _Float16*  hmaxh  = (_Float16*) (ws + 33493504);   //  2,097,152 B

  LAL_knn_kernel<<<(B_ * N_) / 32, 256, 0, stream>>>(xyz, idx);
  LAL_cvt_f16<<<(B_ * N_ * C_) / 256, 256, 0, stream>>>(x, xh);
  LAL_cvt_w1<<<(128 * 512) / 256, 256, 0, stream>>>(w1, wh1);
  LAL_gemm1_kernel<<<(B_ * N_) / 16, 256, 0, stream>>>(xh, wh1, z);
  LAL_stats1_partial<<<S1_BLOCKS, 256, 0, stream>>>(z, idx, part1);
  LAL_stats1_final<<<1, 64, 0, stream>>>(part1, g1, b1, st1);
  LAL_hmax_kernel<<<(B_ * N_) / 4, 256, 0, stream>>>(z, idx, st1, hmax);
  LAL_cvt_f16<<<(B_ * N_ * HID_) / 256, 256, 0, stream>>>(hmax, hmaxh);
  LAL_cvt_f16<<<(C_ * HID_) / 256, 256, 0, stream>>>(w2, w2h);
  LAL_gemm2_kernel<<<dim3((B_ * N_) / 16, 4), 256, 0, stream>>>(hmaxh, w2h, y);
  LAL_stats2_partial<<<(B_ * N_) / 32, 256, 0, stream>>>(y, part2);
  LAL_stats2_final<<<1, 512, 0, stream>>>(part2, g2, b2, st2);
  LAL_apply_bn2<<<(B_ * N_ * C_) / 256, 256, 0, stream>>>(y, st2);
}
